// SEER_ENCODER_26199300505983
// MI455X (gfx1250) — compile-verified, matrix operand order checked by reading, not run on hardware
//
#include <hip/hip_runtime.h>
#include <math.h>

typedef __attribute__((ext_vector_type(16))) _Float16 v16h;
typedef __attribute__((ext_vector_type(8)))  float    v8f;
typedef _Float16 h16;

#define WAVES  8
#define NTHR   256
#define BM     16      // batch items per workgroup
#define ITPW   2       // items per wave
#define DMODEL 128
#define NSEQ   12
#define XSTR   136     // halfs per x row (16B aligned, bank-rotating)
#define WSTR   136     // halfs per weight row
#define QSTR   136     // halfs per Q/K row
#define VSTR   24      // halfs per Vt row
#define WSLOT  (128*WSTR)

// ---- LDS layout (bytes) ----
#define OFF_W   0
#define SZ_W    (4*WSLOT*2)                                  // 139264
#define OFF_X   (OFF_W + SZ_W)
#define SZ_X    (BM*NSEQ*XSTR*2)                             // 52224
#define OFF_S   (OFF_X + SZ_X)
#define SZ_S_PW (16*QSTR*2 + 16*QSTR*2 + 16*16*2 + 128*VSTR*2) // QI+KI+ATh+Vt = 15360
#define SZ_S    (WAVES*SZ_S_PW)                              // 122880
#define LDS_BYTES (OFF_S + SZ_S)                             // 314368 (<320KB)

union AFrag { v16h h; int4 q[2]; };
union H8    { int4 q; h16 h[8]; };

__device__ __forceinline__ v8f wmma_f16(v16h a, v16h b, v8f c) {
  return __builtin_amdgcn_wmma_f32_16x16x32_f16(false, a, false, b, (short)0, c, false, false);
}

// Pack 8 contiguous f32 -> 8 f16 (one 16B LDS store's worth).
__device__ __forceinline__ int4 pack8(const float* __restrict__ src) {
  float4 a = *(const float4*)src;
  float4 b = *(const float4*)(src + 4);
  H8 u;
  u.h[0] = (h16)a.x; u.h[1] = (h16)a.y; u.h[2] = (h16)a.z; u.h[3] = (h16)a.w;
  u.h[4] = (h16)b.x; u.h[5] = (h16)b.y; u.h[6] = (h16)b.z; u.h[7] = (h16)b.w;
  return u.q;
}

// Pack 8 strided f32 (column walk of a row-major matrix) -> 8 f16.
__device__ __forceinline__ int4 pack8_strided(const float* __restrict__ src, int stride) {
  H8 u;
#pragma unroll
  for (int j = 0; j < 8; ++j) u.h[j] = (h16)src[(size_t)j * stride];
  return u.q;
}

// A-style fragment loader from a row-major [rows][stride] f16 buffer, K window = kbase..kbase+31.
// Same loader serves B operands when the buffer holds B^T row-major.
__device__ __forceinline__ v16h ld_frag(const h16* buf, int stride, int kbase) {
  const int lane = threadIdx.x & 31;
  const int r    = lane & 15;
  const int kb   = kbase + ((lane & 16) ? 8 : 0);
  const h16* p   = buf + r * stride + kb;
  AFrag f;
  f.q[0] = *(const int4*)p;
  f.q[1] = *(const int4*)(p + 16);
  return f.h;
}

// Fragment whose K=16..31 slice is identically zero (buffer holds only K=0..15).
__device__ __forceinline__ v16h ld_frag_k16(const h16* buf, int stride) {
  const int lane = threadIdx.x & 31;
  const int r    = lane & 15;
  const int kb   = (lane & 16) ? 8 : 0;
  AFrag f;
  f.q[0] = *(const int4*)(buf + r * stride + kb);
  f.q[1] = make_int4(0, 0, 0, 0);
  return f.h;
}

// x fragment: rows 12..15 are implicit zeros (sequence padded to M=16).
__device__ __forceinline__ v16h ld_frag_x(const h16* xi, int kbase) {
  const int lane = threadIdx.x & 31;
  const int r    = lane & 15;
  const int kb   = kbase + ((lane & 16) ? 8 : 0);
  AFrag f;
  if (r < NSEQ) {
    const h16* p = xi + r * XSTR + kb;
    f.q[0] = *(const int4*)p;
    f.q[1] = *(const int4*)(p + 16);
  } else {
    f.q[0] = make_int4(0, 0, 0, 0);
    f.q[1] = f.q[0];
  }
  return f.h;
}

// Store a 16x16 f32 accumulator tile as f16 into a row-major [16][stride] buffer.
__device__ __forceinline__ void st_tile_half(h16* buf, int stride, int ncol, v8f c) {
  const int lane = threadIdx.x & 31;
  const int n    = lane & 15;
  const int mb   = (lane & 16) ? 8 : 0;
#pragma unroll
  for (int v2 = 0; v2 < 8; ++v2)
    buf[(mb + v2) * stride + ncol + n] = (h16)c[v2];
}

// Branchless tanh-form GELU: one v_exp_f32 + one v_rcp_f32 + a few FMAs.
// Precision is consistent with the f16 WMMA pipeline; Inf-safe at both tails:
// tanh(u) = 1 - 2/(exp(2u)+1), exp->Inf gives tanh->1, exp->0 gives tanh->-1.
__device__ __forceinline__ float gelu_f(float x) {
  float u = 0.7978845608028654f * (x + 0.044715f * x * x * x);
  float t = __expf(2.0f * u);
  float th = 1.0f - 2.0f * __builtin_amdgcn_rcpf(t + 1.0f);
  return 0.5f * x * (1.0f + th);
}

extern "C" __global__ __launch_bounds__(NTHR, 1)
void seer_encoder_fused(const float* __restrict__ obs,
                        const float* __restrict__ task,
                        const float* __restrict__ te_w1, const float* __restrict__ te_b1,
                        const float* __restrict__ te_w2, const float* __restrict__ te_b2,
                        const float* __restrict__ te_w3, const float* __restrict__ te_b3,
                        const float* __restrict__ action_token,
                        const float* __restrict__ pred_token,
                        const float* __restrict__ qkv_w, const float* __restrict__ qkv_b,
                        const float* __restrict__ lin_w, const float* __restrict__ lin_b,
                        float* __restrict__ out, int Btot)
{
  __shared__ __align__(16) unsigned char lds[LDS_BYTES];
  h16* sW = (h16*)(lds + OFF_W);          // 4 x [128][WSTR]  (W^T, f16)
  h16* sX = (h16*)(lds + OFF_X);          // [BM][NSEQ][XSTR] (x, f16)
  unsigned char* sS = lds + OFF_S;        // per-wave scratch

  const int tid   = threadIdx.x;
  const int wave  = tid >> 5;
  const int lane  = tid & 31;
  const int n     = lane & 15;
  const int hi    = (lane >> 4) & 1;
  const int bbase = blockIdx.x * BM;

  // staging role: each thread owns transposed-weight row `wn`, k-half `wk0`
  const int wn  = tid & 127;
  const int wk0 = (tid >> 7) * 64;

  h16* QI  = (h16*)(sS + wave * SZ_S_PW); // [16][QSTR]
  h16* KI  = QI + 16 * QSTR;              // [16][QSTR]
  h16* ATh = KI + 16 * QSTR;              // [16][16]
  h16* Vt  = ATh + 16 * 16;               // [128][VSTR] : V transposed (d,k)

  // Phase-1 overlays in the (not yet used) scratch region
  h16* sTask = (h16*)sS;                  // [16][768]
  h16* sH1   = (h16*)(sS + 16 * 768 * 2); // [16][QSTR]
  h16* sH2   = sH1 + 16 * QSTR;           // [16][QSTR]

  // ---------------- Phase 0: stage inputs (vectorized fp32 -> f16) ----------------
  for (int idx = tid; idx < BM * 96; idx += NTHR) {            // task: 16 x 768
    int i = idx / 96, kb = (idx - i * 96) * 8;
    *(int4*)(sTask + i * 768 + kb) = pack8(task + (size_t)(bbase + i) * 768 + kb);
  }
  for (int idx = tid; idx < BM * 5 * 16; idx += NTHR) {        // obs: 16 x 5 x 128
    int i = idx / 80, r = idx - i * 80;
    int t = r >> 4, d8 = (r & 15) * 8;
    *(int4*)(sX + (i * NSEQ + t) * XSTR + d8) =
        pack8(obs + ((size_t)(bbase + i) * 5 + t) * DMODEL + d8);
  }
  for (int idx = tid; idx < BM * 6 * 16; idx += NTHR) {        // action (row6) + pred (rows7..11)
    int i = idx / 96, r = idx - i * 96;
    int t = r >> 4, d8 = (r & 15) * 8;
    const float* src = (t == 0) ? (action_token + d8) : (pred_token + (t - 1) * DMODEL + d8);
    *(int4*)(sX + (i * NSEQ + 6 + t) * XSTR + d8) = pack8(src);
  }
  __syncthreads();

  // ---------------- Phase 1: task-embedding MLP (M=16 items, wave w owns N-tile w) ----
  {
    v8f acc = {};
    for (int c = 0; c < 6; ++c) {          // K = 768 in six 128-chunks
      __syncthreads();
      const float* src = te_w1 + (size_t)c * 128 * 128;
#pragma unroll
      for (int k8 = 0; k8 < 8; ++k8) {
        int kb = wk0 + k8 * 8;
        *(int4*)(sW + wn * WSTR + kb) = pack8_strided(src + (size_t)kb * 128 + wn, 128);
      }
      __syncthreads();
#pragma unroll
      for (int ks = 0; ks < 4; ++ks)
        acc = wmma_f16(ld_frag(sTask, 768, c * 128 + ks * 32),
                       ld_frag(sW + wave * 16 * WSTR, WSTR, ks * 32), acc);
    }
    float bb = te_b1[wave * 16 + n];
#pragma unroll
    for (int v2 = 0; v2 < 8; ++v2) acc[v2] = gelu_f(acc[v2] + bb);
    st_tile_half(sH1, QSTR, wave * 16, acc);
  }
  __syncthreads();
  {
#pragma unroll
    for (int k8 = 0; k8 < 8; ++k8) {
      int kb = wk0 + k8 * 8;
      *(int4*)(sW + wn * WSTR + kb) = pack8_strided(te_w2 + (size_t)kb * 128 + wn, 128);
    }
    __syncthreads();
    v8f acc = {};
#pragma unroll
    for (int ks = 0; ks < 4; ++ks)
      acc = wmma_f16(ld_frag(sH1, QSTR, ks * 32),
                     ld_frag(sW + wave * 16 * WSTR, WSTR, ks * 32), acc);
    float bb = te_b2[wave * 16 + n];
#pragma unroll
    for (int v2 = 0; v2 < 8; ++v2) acc[v2] = gelu_f(acc[v2] + bb);
    st_tile_half(sH2, QSTR, wave * 16, acc);
  }
  __syncthreads();
  {
#pragma unroll
    for (int k8 = 0; k8 < 8; ++k8) {
      int kb = wk0 + k8 * 8;
      *(int4*)(sW + wn * WSTR + kb) = pack8_strided(te_w3 + (size_t)kb * 128 + wn, 128);
    }
    __syncthreads();
    v8f acc = {};
#pragma unroll
    for (int ks = 0; ks < 4; ++ks)
      acc = wmma_f16(ld_frag(sH2, QSTR, ks * 32),
                     ld_frag(sW + wave * 16 * WSTR, WSTR, ks * 32), acc);
    float bb = te_b3[wave * 16 + n];
#pragma unroll
    for (int v2 = 0; v2 < 8; ++v2) {
      int m = hi * 8 + v2;                 // m = item index
      sX[(m * NSEQ + 5) * XSTR + wave * 16 + n] = (h16)(acc[v2] + bb);  // task_z -> token 5
    }
  }
  __syncthreads();

  // ---------------- Phase 2: three attention layers ----------------
  for (int l = 0; l < 3; ++l) {
    // Hoist this layer's per-lane bias values into registers (one load clause,
    // latency hidden under weight staging; reused for both items).
    float bq[8], bk[8], bv[8], bl[8];
#pragma unroll
    for (int t8 = 0; t8 < 8; ++t8) {
      bq[t8] = qkv_b[((size_t)l * 3 + 0) * 128 + t8 * 16 + n];
      bk[t8] = qkv_b[((size_t)l * 3 + 1) * 128 + t8 * 16 + n];
      bv[t8] = qkv_b[((size_t)l * 3 + 2) * 128 + t8 * 16 + n];
      bl[t8] = lin_b[(size_t)l * 128 + t8 * 16 + n];
    }

    // Stage Wq^T, Wk^T, Wv^T, Wlin^T as f16 (vectorized transpose staging)
#pragma unroll
    for (int mat = 0; mat < 4; ++mat) {
      const float* src = (mat < 3) ? (qkv_w + (((size_t)l * 3 + mat) << 14))
                                   : (lin_w + ((size_t)l << 14));
      h16* dst = sW + mat * WSLOT + wn * WSTR;
#pragma unroll
      for (int k8 = 0; k8 < 8; ++k8) {
        int kb = wk0 + k8 * 8;
        *(int4*)(dst + kb) = pack8_strided(src + (size_t)kb * 128 + wn, 128);
      }
    }
    __syncthreads();

    for (int rr = 0; rr < ITPW; ++rr) {
      const int it = wave * ITPW + rr;
      h16* xi = sX + it * NSEQ * XSTR;

      v16h ax[4];
#pragma unroll
      for (int ks = 0; ks < 4; ++ks) ax[ks] = ld_frag_x(xi, ks * 32);

      // ---- Q, K projections (each: 8 N-tiles x 4 K-steps of WMMA) ----
#pragma unroll
      for (int j = 0; j < 2; ++j) {
        h16* dst = j ? KI : QI;
        const h16* wm = sW + j * WSLOT;
#pragma unroll
        for (int t8 = 0; t8 < 8; ++t8) {
          v8f acc = {};
#pragma unroll
          for (int ks = 0; ks < 4; ++ks)
            acc = wmma_f16(ax[ks], ld_frag(wm + t8 * 16 * WSTR, WSTR, ks * 32), acc);
          float bb = j ? bk[t8] : bq[t8];
#pragma unroll
          for (int v2 = 0; v2 < 8; ++v2) acc[v2] += bb;
          st_tile_half(dst, QSTR, t8 * 16, acc);
        }
      }

      // ---- qk = Q @ K^T  (B operand = K rows via the A-style loader) ----
      v8f qk = {};
#pragma unroll
      for (int ks = 0; ks < 4; ++ks)
        qk = wmma_f16(ld_frag(QI, QSTR, ks * 32), ld_frag(KI, QSTR, ks * 32), qk);

      // ---- masked softmax over rows; masked/pad columns become exact 0 ----
#pragma unroll
      for (int v2 = 0; v2 < 8; ++v2) {
        int m = hi * 8 + v2;
        bool masked = (n >= 7) || (n == 6 && m < 7) || (n < 5 && m < 5 && n > m);
        float val  = masked ? -__builtin_inff() : qk[v2];
        float rmax = val;
#pragma unroll
        for (int off = 8; off > 0; off >>= 1)
          rmax = fmaxf(rmax, __shfl_xor(rmax, off, 16));
        float e = masked ? 0.0f : __expf(val - rmax);
        float ssum = e;
#pragma unroll
        for (int off = 8; off > 0; off >>= 1)
          ssum += __shfl_xor(ssum, off, 16);
        ATh[m * 16 + n] = (h16)(e * __builtin_amdgcn_rcpf(ssum));
      }

      // ---- V projection, stored transposed: Vt[d][k_token] ----
      {
        const h16* wm = sW + 2 * WSLOT;
#pragma unroll
        for (int t8 = 0; t8 < 8; ++t8) {
          v8f acc = {};
#pragma unroll
          for (int ks = 0; ks < 4; ++ks)
            acc = wmma_f16(ax[ks], ld_frag(wm + t8 * 16 * WSTR, WSTR, ks * 32), acc);
          float bb = bv[t8];
#pragma unroll
          for (int v2 = 0; v2 < 8; ++v2)
            Vt[(t8 * 16 + n) * VSTR + hi * 8 + v2] = (h16)(acc[v2] + bb);
        }
      }

      // ---- attnV = attn @ V : single K=32 WMMA per N-tile (upper K half is reg-zero) ----
      {
        v16h ap = ld_frag_k16(ATh, 16);
#pragma unroll
        for (int t8 = 0; t8 < 8; ++t8) {
          v8f acc = {};
          acc = wmma_f16(ap, ld_frag_k16(Vt + t8 * 16 * VSTR, VSTR), acc);
          st_tile_half(QI, QSTR, t8 * 16, acc);   // reuse QI (Q is dead)
        }
      }

      // ---- x = gelu(attnV @ lin_w + lin_b) ; final layer streams rows 6..11 out ----
      {
        v16h aa[4];
#pragma unroll
        for (int ks = 0; ks < 4; ++ks) aa[ks] = ld_frag(QI, QSTR, ks * 32);
        const h16* wm = sW + 3 * WSLOT;
#pragma unroll
        for (int t8 = 0; t8 < 8; ++t8) {
          v8f acc = {};
#pragma unroll
          for (int ks = 0; ks < 4; ++ks)
            acc = wmma_f16(aa[ks], ld_frag(wm + t8 * 16 * WSTR, WSTR, ks * 32), acc);
          float bb = bl[t8];
          if (l < 2) {
#pragma unroll
            for (int v2 = 0; v2 < 8; ++v2) {
              int m = hi * 8 + v2;
              if (m < NSEQ)                               // gelu only for stored rows
                xi[m * XSTR + t8 * 16 + n] = (h16)gelu_f(acc[v2] + bb);
            }
          } else {
            int itemg = bbase + it;
#pragma unroll
            for (int v2 = 0; v2 < 8; ++v2) {
              int m = hi * 8 + v2;
              if (m >= 6 && m < 12) {                     // only rows 6..11 leave the chip
                float g = gelu_f(acc[v2] + bb);
                if (m == 6)
                  out[(size_t)itemg * 128 + t8 * 16 + n] = g;                       // action_z
                else
                  out[(size_t)Btot * 128 + ((size_t)itemg * 5 + (m - 7)) * 128 + t8 * 16 + n] = g; // pred_z
              }
            }
          }
        }
      }
    } // rr
    __syncthreads();
  } // layers
}

extern "C" void kernel_launch(void* const* d_in, const int* in_sizes, int n_in,
                              void* d_out, int out_size, void* d_ws, size_t ws_size,
                              hipStream_t stream) {
  (void)n_in; (void)out_size; (void)d_ws; (void)ws_size;
  const float* obs   = (const float*)d_in[0];
  const float* task  = (const float*)d_in[1];
  const float* te_w1 = (const float*)d_in[2];
  const float* te_b1 = (const float*)d_in[3];
  const float* te_w2 = (const float*)d_in[4];
  const float* te_b2 = (const float*)d_in[5];
  const float* te_w3 = (const float*)d_in[6];
  const float* te_b3 = (const float*)d_in[7];
  const float* atok  = (const float*)d_in[8];
  const float* ptok  = (const float*)d_in[9];
  const float* qkv_w = (const float*)d_in[10];
  const float* qkv_b = (const float*)d_in[11];
  const float* lin_w = (const float*)d_in[12];
  const float* lin_b = (const float*)d_in[13];

  int Btot = in_sizes[0] / (5 * 128);   // obs is (B, 5, 128)
  dim3 grid(Btot / BM), block(NTHR);
  hipLaunchKernelGGL(seer_encoder_fused, grid, block, 0, stream,
                     obs, task, te_w1, te_b1, te_w2, te_b2, te_w3, te_b3,
                     atok, ptok, qkv_w, qkv_b, lin_w, lin_b, (float*)d_out, Btot);
}
